// MIWAE_43731357008597
// MI455X (gfx1250) — matrix operand order repaired and checked by reading, not compiled, not run on hardware
//
#include <hip/hip_runtime.h>
#include <math.h>

// ---------------- problem constants ----------------
#define BB 2048
#define PP 1024
#define HH 2048
#define DD 256
#define LL 10
#define NCLS 10
#define LOG2PI 1.8378770664093453f
#define TWOPI 6.283185307179586f

typedef __attribute__((ext_vector_type(16))) _Float16 v16h;
typedef __attribute__((ext_vector_type(8)))  float    v8f;

// ---------------- GEMM: C = [relu](A @ B + bias) ----------------
// A: MxK row-major (fp32), B: KxN row-major (fp32), C: MxN row-major (fp32)
// Requires: M % 128 == 0, K % 32 == 0, lda % 4 == 0.
// ALIGNED_N: N % 128 == 0 (fast path). Otherwise ragged-N guards.
#define BLK_M 128
#define BLK_N 128
#define BLK_K 32
#define AS_LD 34
#define BS_LD 34

template <bool RELU, bool ALIGNED_N>
__global__ __launch_bounds__(256) void gemm_bias(
    const float* __restrict__ A, const float* __restrict__ Bw,
    const float* __restrict__ bias, float* __restrict__ C,
    int M, int N, int K, int lda, int ldb, int ldc) {
  __shared__ _Float16 As[BLK_M * AS_LD];
  __shared__ _Float16 Bs[BLK_N * BS_LD];

  const int tid  = threadIdx.x;
  const int lane = tid & 31;
  const int wave = tid >> 5;
  const int wm   = wave >> 1;  // 4 row slots of 32
  const int wn   = wave & 1;   // 2 col slots of 64
  const int g    = lane >> 4;  // lane group within wave
  const int l16  = lane & 15;

  const int m0 = blockIdx.y * BLK_M;
  const int n0 = blockIdx.x * BLK_N;

  // loader thread mapping
  const int ar  = tid >> 3;         // A: row 0..31 (+32*pass)
  const int ac4 = (tid & 7) * 4;    // A: col group
  const int bnl = tid & 127;        // B: local column
  const int bkb = (tid >> 7) * 16;  // B: k base (0 or 16)
  const bool bok = ALIGNED_N || (n0 + bnl < N);

  v8f acc[2][4];
  const v8f vzero = {0.f, 0.f, 0.f, 0.f, 0.f, 0.f, 0.f, 0.f};
#pragma unroll
  for (int i = 0; i < 2; ++i)
#pragma unroll
    for (int j = 0; j < 4; ++j) acc[i][j] = vzero;

  union U16 { v16h h; unsigned int u[8]; };

  float4 pa[4];
  float  pb[16];

  // prefetch k0 = 0 into registers
#pragma unroll
  for (int t = 0; t < 4; ++t)
    pa[t] = *reinterpret_cast<const float4*>(
        A + (size_t)(m0 + ar + 32 * t) * lda + ac4);
#pragma unroll
  for (int j = 0; j < 16; ++j)
    pb[j] = bok ? Bw[(size_t)(bkb + j) * ldb + (n0 + bnl)] : 0.f;

  for (int k0 = 0; k0 < K; k0 += BLK_K) {
    // ---- commit staged registers to LDS ----
#pragma unroll
    for (int t = 0; t < 4; ++t) {
      _Float16* d = &As[(ar + 32 * t) * AS_LD + ac4];
      d[0] = (_Float16)pa[t].x; d[1] = (_Float16)pa[t].y;
      d[2] = (_Float16)pa[t].z; d[3] = (_Float16)pa[t].w;
    }
    {
      _Float16* d = &Bs[bnl * BS_LD + bkb];
#pragma unroll
      for (int j = 0; j < 16; ++j) d[j] = (_Float16)pb[j];
    }
    __syncthreads();

    // ---- prefetch next tile while WMMAs run ----
    const int kn = k0 + BLK_K;
    if (kn < K) {
#pragma unroll
      for (int t = 0; t < 4; ++t)
        pa[t] = *reinterpret_cast<const float4*>(
            A + (size_t)(m0 + ar + 32 * t) * lda + kn + ac4);
#pragma unroll
      for (int j = 0; j < 16; ++j)
        pb[j] = bok ? Bw[(size_t)(kn + bkb + j) * ldb + (n0 + bnl)] : 0.f;
    }

    // ---- build fragments per CDNA5 16-bit layouts, issue WMMA ----
    U16 afrag[2], bfrag[4];
#pragma unroll
    for (int mt = 0; mt < 2; ++mt) {
      const int mrow = wm * 32 + mt * 16 + l16;
#pragma unroll
      for (int p = 0; p < 8; ++p) {
        const int kk = 2 * p + 8 * g + ((p >= 4) ? 8 : 0);  // A: K pairs
        afrag[mt].u[p] =
            *reinterpret_cast<const unsigned int*>(&As[mrow * AS_LD + kk]);
      }
    }
#pragma unroll
    for (int nt = 0; nt < 4; ++nt) {
      const int nrow = wn * 64 + nt * 16 + l16;
#pragma unroll
      for (int p = 0; p < 8; ++p) {
        const int kk = 2 * p + 8 * g + ((p >= 4) ? 8 : 0);  // B: K pairs
        bfrag[nt].u[p] =
            *reinterpret_cast<const unsigned int*>(&Bs[nrow * BS_LD + kk]);
      }
    }
#pragma unroll
    for (int mt = 0; mt < 2; ++mt)
#pragma unroll
      for (int nt = 0; nt < 4; ++nt)
        acc[mt][nt] = __builtin_amdgcn_wmma_f32_16x16x32_f16(
            false, afrag[mt].h, false, bfrag[nt].h, (short)0, acc[mt][nt],
            false, false);
    __syncthreads();
  }

  // ---- epilogue: bias (+relu), C layout: VGPR r -> M = r + 8*g ----
#pragma unroll
  for (int mt = 0; mt < 2; ++mt)
#pragma unroll
    for (int nt = 0; nt < 4; ++nt) {
      const int col = n0 + wn * 64 + nt * 16 + l16;
      if (ALIGNED_N || col < N) {
        const float bv = bias[col];
#pragma unroll
        for (int r = 0; r < 8; ++r) {
          const int row = m0 + wm * 32 + mt * 16 + r + 8 * g;
          float v = acc[mt][nt][r] + bv;
          if (RELU) v = fmaxf(v, 0.f);
          C[(size_t)row * ldc + col] = v;
        }
      }
    }
}

// ---------------- helpers ----------------
__device__ inline unsigned int pcg(unsigned int v) {
  unsigned int s = v * 747796405u + 2891336453u;
  unsigned int w = ((s >> ((s >> 28) + 4u)) ^ s) * 277803737u;
  return (w >> 22) ^ w;
}
__device__ inline float to01(unsigned int h) {
  return ((float)h + 0.5f) * 2.3283064365386963e-10f;  // (0,1)
}
__device__ inline float softplusf(float x) {
  return (x > 15.f) ? x : log1pf(expf(x));
}
__device__ inline float block_reduce_sum(float v, float* sbuf) {
  const int tid = threadIdx.x;
  sbuf[tid] = v;
  __syncthreads();
  for (int s = 128; s > 0; s >>= 1) {
    if (tid < s) sbuf[tid] += sbuf[tid + s];
    __syncthreads();
  }
  const float r = sbuf[0];
  __syncthreads();
  return r;
}

// ---------------- mu / sigma split ----------------
__global__ __launch_bounds__(256) void k_musig(const float* __restrict__ oe,
                                               float* __restrict__ mu,
                                               float* __restrict__ sig) {
  const int i = blockIdx.x * 256 + threadIdx.x;  // < BB*DD
  const int b = i / DD, d = i % DD;
  mu[i]  = oe[(size_t)b * (2 * DD) + d];
  sig[i] = softplusf(oe[(size_t)b * (2 * DD) + DD + d]);  // +1e-64 ~ 0 in f32
}

// ---------------- z = mu + sig*eps, logpz, logq ----------------
__global__ __launch_bounds__(256) void k_sample(
    const float* __restrict__ mu, const float* __restrict__ sig,
    float* __restrict__ z, float* __restrict__ logpz,
    float* __restrict__ logq) {
  __shared__ float sbuf[256];
  const int row = blockIdx.x;        // l*BB + b
  const int b   = row % BB;
  const int d   = threadIdx.x;       // DD == 256
  const unsigned int idx = (unsigned int)(row * DD + d);
  const float u1 = to01(pcg(idx * 2u + 0x9E3779B9u));
  const float u2 = to01(pcg(idx * 2u + 1u + 0x9E3779B9u));
  const float eps = sqrtf(-2.f * logf(u1)) * cosf(TWOPI * u2);
  const float m = mu[b * DD + d], s = sig[b * DD + d];
  const float zv = m + s * eps;
  z[(size_t)row * DD + d] = zv;
  const float s1 = block_reduce_sum(zv * zv, sbuf);
  const float s2 = block_reduce_sum(-0.5f * eps * eps - logf(s), sbuf);
  if (d == 0) {
    logpz[row] = -0.5f * s1 - 0.5f * (float)DD * LOG2PI;
    logq[row]  = s2 - 0.5f * (float)DD * LOG2PI;
  }
}

// ---------------- decoder epilogue: student-t logprob + xms sample ----------
__global__ __launch_bounds__(256) void k_dec_epi(
    const float* __restrict__ od, const float* __restrict__ data,
    const int* __restrict__ mask, float* __restrict__ xms,
    float* __restrict__ logpxobs, int l) {
  __shared__ float sbuf[256];
  const int b = blockIdx.x;
  const float* row = od + (size_t)b * (3 * PP);
  float lp = 0.f;
#pragma unroll
  for (int j = 0; j < 4; ++j) {
    const int p = threadIdx.x + j * 256;
    const float mean  = row[p];
    const float scale = softplusf(row[PP + p]) + 0.001f;
    const float df    = softplusf(row[2 * PP + p]) + 3.0f;
    const float x = data[(size_t)b * PP + p];
    const float y = (x - mean) / scale;
    const float lpx = lgammaf(0.5f * (df + 1.f)) - lgammaf(0.5f * df)
                    - 0.5f * logf(df * 3.14159265358979f) - logf(scale)
                    - 0.5f * (df + 1.f) * log1pf(y * y / df);
    lp += mask[(size_t)b * PP + p] ? lpx : 0.f;
    // Bailey's polar method: exact student-t(df) sample
    const unsigned int i = (unsigned int)(((l * BB) + b) * PP + p);
    const float u1 = to01(pcg(i * 2u + 0x85EBCA6Bu));
    const float u2 = to01(pcg(i * 2u + 1u + 0x85EBCA6Bu));
    const float t = sqrtf(df * (expf((-2.f / df) * logf(u1)) - 1.f)) *
                    cosf(TWOPI * u2);
    xms[((size_t)l * BB + b) * PP + p] = mean + scale * t;
  }
  const float s = block_reduce_sum(lp, sbuf);
  if (threadIdx.x == 0) logpxobs[l * BB + b] = s;
}

// ---------------- importance weights: softmax over L ----------------
__global__ __launch_bounds__(256) void k_impw(const float* __restrict__ lpx,
                                              const float* __restrict__ lpz,
                                              const float* __restrict__ lq,
                                              float* __restrict__ w) {
  const int b = blockIdx.x * 256 + threadIdx.x;
  if (b >= BB) return;
  float v[LL], mx = -3.4e38f;
#pragma unroll
  for (int l = 0; l < LL; ++l) {
    v[l] = lpx[l * BB + b] + lpz[l * BB + b] - lq[l * BB + b];
    mx = fmaxf(mx, v[l]);
  }
  float s = 0.f;
#pragma unroll
  for (int l = 0; l < LL; ++l) { v[l] = expf(v[l] - mx); s += v[l]; }
  const float inv = 1.f / s;
#pragma unroll
  for (int l = 0; l < LL; ++l) w[l * BB + b] = v[l] * inv;
}

// ---------------- imputation ----------------
__global__ __launch_bounds__(256) void k_impute(
    const float* __restrict__ xms, const float* __restrict__ w,
    const float* __restrict__ data, const int* __restrict__ mask,
    float* __restrict__ out) {
  const size_t i = (size_t)blockIdx.x * 256 + threadIdx.x;  // < BB*PP
  const int b = (int)(i >> 10), p = (int)(i & 1023);
  float acc = 0.f;
#pragma unroll
  for (int l = 0; l < LL; ++l)
    acc += w[l * BB + b] * xms[((size_t)l * BB + b) * PP + p];
  out[i] = mask[i] ? data[i] : acc;
}

// ---------------- host orchestration ----------------
extern "C" void kernel_launch(void* const* d_in, const int* in_sizes, int n_in,
                              void* d_out, int out_size, void* d_ws,
                              size_t ws_size, hipStream_t stream) {
  const float* data   = (const float*)d_in[0];
  const int*   mask   = (const int*)d_in[1];
  const float* enc_w1 = (const float*)d_in[2];
  const float* enc_b1 = (const float*)d_in[3];
  const float* enc_w2 = (const float*)d_in[4];
  const float* enc_b2 = (const float*)d_in[5];
  const float* dec_w1 = (const float*)d_in[6];
  const float* dec_b1 = (const float*)d_in[7];
  const float* dec_w2 = (const float*)d_in[8];
  const float* dec_b2 = (const float*)d_in[9];
  const float* cls_w1 = (const float*)d_in[10];
  const float* cls_b1 = (const float*)d_in[11];
  const float* cls_w2 = (const float*)d_in[12];
  const float* cls_b2 = (const float*)d_in[13];
  float* out = (float*)d_out;

  // workspace layout (~170 MB, all chunks 16B-aligned)
  float* ws       = (float*)d_ws;
  float* mu       = ws;                         // BB*DD
  float* sig      = mu + (size_t)BB * DD;       // BB*DD
  float* z        = sig + (size_t)BB * DD;      // LL*BB*DD
  float* logpz    = z + (size_t)LL * BB * DD;   // LL*BB
  float* logq     = logpz + (size_t)LL * BB;    // LL*BB
  float* logpxobs = logq + (size_t)LL * BB;     // LL*BB
  float* impw     = logpxobs + (size_t)LL * BB; // LL*BB
  float* he       = impw + (size_t)LL * BB;     // BB*HH (also hc)
  float* out_enc  = he + (size_t)BB * HH;       // BB*2DD
  float* hd       = out_enc + (size_t)BB * 2 * DD;  // BB*HH (per-l chunk)
  float* out_dec  = hd + (size_t)BB * HH;       // BB*3PP (per-l chunk)
  float* xms      = out_dec + (size_t)BB * 3 * PP;  // LL*BB*PP

  const dim3 blk(256);

  // encoder
  gemm_bias<true, true><<<dim3(HH / BLK_N, BB / BLK_M), blk, 0, stream>>>(
      data, enc_w1, enc_b1, he, BB, HH, PP, PP, HH, HH);
  gemm_bias<false, true><<<dim3((2 * DD) / BLK_N, BB / BLK_M), blk, 0,
                           stream>>>(he, enc_w2, enc_b2, out_enc, BB, 2 * DD,
                                     HH, HH, 2 * DD, 2 * DD);
  k_musig<<<(BB * DD) / 256, blk, 0, stream>>>(out_enc, mu, sig);
  k_sample<<<LL * BB, blk, 0, stream>>>(mu, sig, z, logpz, logq);

  // decoder, chunked over importance samples
  for (int l = 0; l < LL; ++l) {
    gemm_bias<true, true><<<dim3(HH / BLK_N, BB / BLK_M), blk, 0, stream>>>(
        z + (size_t)l * BB * DD, dec_w1, dec_b1, hd, BB, HH, DD, DD, HH, HH);
    gemm_bias<false, true><<<dim3((3 * PP) / BLK_N, BB / BLK_M), blk, 0,
                             stream>>>(hd, dec_w2, dec_b2, out_dec, BB, 3 * PP,
                                       HH, HH, 3 * PP, 3 * PP);
    k_dec_epi<<<BB, blk, 0, stream>>>(out_dec, data, mask, xms, logpxobs, l);
  }

  // weights + imputation
  k_impw<<<(BB + 255) / 256, blk, 0, stream>>>(logpxobs, logpz, logq, impw);
  k_impute<<<((size_t)BB * PP) / 256, blk, 0, stream>>>(xms, impw, data, mask,
                                                        out);

  // classifier on imputed data (reads d_out region 0..BB*PP)
  gemm_bias<true, true><<<dim3(HH / BLK_N, BB / BLK_M), blk, 0, stream>>>(
      out, cls_w1, cls_b1, he, BB, HH, PP, PP, HH, HH);
  gemm_bias<false, false><<<dim3(1, BB / BLK_M), blk, 0, stream>>>(
      he, cls_w2, cls_b2, out + (size_t)BB * PP, BB, NCLS, HH, HH, NCLS, NCLS);
}
